// RoPEMultiheadAttention_52080773432041
// MI455X (gfx1250) — matrix operand order repaired and checked by reading, not compiled, not run on hardware
//
#include <hip/hip_runtime.h>
#include <hip/hip_bf16.h>
#include <stdint.h>

typedef __attribute__((ext_vector_type(16))) _Float16 v16h;
typedef __attribute__((ext_vector_type(8)))  _Float16 v8h;
typedef __attribute__((ext_vector_type(8)))  float    v8f;
typedef __attribute__((ext_vector_type(4)))  float    v4f;
typedef __attribute__((ext_vector_type(4)))  uint32_t u32x4;
typedef __attribute__((ext_vector_type(8)))  int      i32x8;
typedef __attribute__((ext_vector_type(4)))  int      i32x4;

#define L_DIM 2048
#define S_DIM 2048
#define B_DIM 2
#define E_DIM 1024
#define H_DIM 16
#define HD    64
#define MROWS (L_DIM * B_DIM)   // 4096

#if __has_builtin(__builtin_amdgcn_tensor_load_to_lds)
#define USE_TDM 1
#else
#define USE_TDM 0
#endif

__device__ __forceinline__ v16h cat8(v8h lo, v8h hi) {
    return __builtin_shufflevector(lo, hi, 0,1,2,3,4,5,6,7,8,9,10,11,12,13,14,15);
}

#if USE_TDM
// TDM 2D tile load: tile_d1 rows x tile_d0 f16 elements from a row-major tensor
// (row stride = stride0 elements), packed contiguously into LDS.
// D# layout per CDNA5 ISA §8.3-8.4; 6-arg builtin (clang-23 form).
__device__ __forceinline__ void tdm_load_2d(uint32_t lds_byte_addr,
                                            const _Float16* gptr,
                                            uint32_t tensor_d0, uint32_t tensor_d1,
                                            uint32_t tile_d0,   uint32_t tile_d1,
                                            uint32_t stride0) {
    const uint64_t ga = (uint64_t)(uintptr_t)gptr;
    u32x4 g0;
    g0[0] = 1u;                                            // count=1, user mode
    g0[1] = lds_byte_addr;                                 // lds_addr
    g0[2] = (uint32_t)ga;                                  // global_addr[31:0]
    g0[3] = (uint32_t)((ga >> 32) & 0x01FFFFFFu) | (2u << 30); // addr[56:32] | type=2
    i32x8 g1;
    g1[0] = 0x00010000;                                    // data_size=1 (2 bytes)
    g1[1] = (int)((tensor_d0 & 0xFFFFu) << 16);            // tensor_dim0 lo
    g1[2] = (int)((tensor_d0 >> 16) | ((tensor_d1 & 0xFFFFu) << 16));
    g1[3] = (int)((tensor_d1 >> 16) | (tile_d0 << 16));    // tensor_dim1 hi | tile_dim0
    g1[4] = (int)(tile_d1 & 0xFFFFu);                      // tile_dim1, tile_dim2=0
    g1[5] = (int)stride0;                                  // tensor_dim0_stride lo
    g1[6] = 0;
    g1[7] = 0;
    const i32x4 z4 = {0, 0, 0, 0};
    const i32x8 z8 = {0, 0, 0, 0, 0, 0, 0, 0};
    __builtin_amdgcn_tensor_load_to_lds(g0, g1, z4, z4, z8, 0);
}
#define TDM_WAIT() __builtin_amdgcn_s_wait_tensorcnt(0)
#endif

// ---------------------------------------------------------------------------
// fp32 -> fp16 bulk conversion
// ---------------------------------------------------------------------------
__global__ __launch_bounds__(256) void cvt_f32_f16_kernel(
    const float* __restrict__ in, _Float16* __restrict__ out)
{
    const size_t i = ((size_t)blockIdx.x * 256 + threadIdx.x) * 8;
    v4f a = *(const v4f*)(in + i);
    v4f b = *(const v4f*)(in + i + 4);
    v8h o;
    #pragma unroll
    for (int j = 0; j < 4; ++j) { o[j] = (_Float16)a[j]; o[4 + j] = (_Float16)b[j]; }
    *(v8h*)(out + i) = o;
}

// ---------------------------------------------------------------------------
// Block-tiled f16 GEMM with double-buffered TDM staging.
//   C[128x128] = A[128,1024] x W^T[1024,128]
//   8 waves as 4(M)x2(N); each wave owns 32x64 (2x4 WMMA accumulators).
// ---------------------------------------------------------------------------
__device__ __forceinline__ void gemm_compute_slice(
    const _Float16* __restrict__ Ab, const _Float16* __restrict__ Bb,
    int wm, int wn, int half, int ln, v8f acc[2][4])
{
    v16h af[2], bf[4];
    #pragma unroll
    for (int mi = 0; mi < 2; ++mi) {
        const _Float16* p = &Ab[(wm * 32 + mi * 16 + ln) * 32 + half * 16];
        af[mi] = *(const v16h*)p;
    }
    #pragma unroll
    for (int nj = 0; nj < 4; ++nj) {
        const _Float16* p = &Bb[(wn * 64 + nj * 16 + ln) * 32 + half * 16];
        bf[nj] = *(const v16h*)p;
    }
    #pragma unroll
    for (int mi = 0; mi < 2; ++mi)
        #pragma unroll
        for (int nj = 0; nj < 4; ++nj)
            acc[mi][nj] = __builtin_amdgcn_wmma_f32_16x16x32_f16(
                false, af[mi], false, bf[nj], (short)0, acc[mi][nj], false, false);
}

// mode 0: Q +RoPE +1/8 -> [bh][l][d]; 1: K +RoPE -> [bh][s][d];
// mode 2: V transposed -> [bh][d][s]; 3: fp32+bias -> outf[m][n]
__global__ __launch_bounds__(256) void gemm_kernel(
    const _Float16* __restrict__ A,     // [4096, 1024] f16
    const _Float16* __restrict__ W,     // [1024, 1024] f16
    const float*    __restrict__ bias,  // [1024]
    _Float16*       __restrict__ outh,
    float*          __restrict__ outf,
    int mode)
{
    __shared__ __align__(64) _Float16 Alds[2][128 * 32];
    __shared__ __align__(64) _Float16 Blds[2][128 * 32];

    const int lane = threadIdx.x & 31;
    const int wv   = threadIdx.x >> 5;
    const int wm   = wv >> 1;
    const int wn   = wv & 1;
    const int half = lane >> 4;
    const int ln   = lane & 15;
    const int m0   = blockIdx.x * 128;
    const int n0   = blockIdx.y * 128;

    const _Float16* Abase = A + (size_t)m0 * E_DIM;
    const _Float16* Bbase = W + (size_t)n0 * E_DIM;
    v8f acc[2][4] = {};

#if USE_TDM
    if (wv == 0) {   // prologue: fill buffer 0 (EXEC ignored by TDM)
        tdm_load_2d((uint32_t)(uintptr_t)&Alds[0][0], Abase, E_DIM, MROWS, 32, 128, E_DIM);
        tdm_load_2d((uint32_t)(uintptr_t)&Blds[0][0], Bbase, E_DIM, E_DIM, 32, 128, E_DIM);
    }
    for (int kt = 0; kt < E_DIM; kt += 64) {
        if (wv == 0) TDM_WAIT();            // buffer 0 slice landed
        __syncthreads();                    // publish; prior readers of buf1 done
        if (wv == 0) {                      // prefetch next slice into buffer 1
            tdm_load_2d((uint32_t)(uintptr_t)&Alds[1][0], Abase + kt + 32, E_DIM, MROWS, 32, 128, E_DIM);
            tdm_load_2d((uint32_t)(uintptr_t)&Blds[1][0], Bbase + kt + 32, E_DIM, E_DIM, 32, 128, E_DIM);
        }
        gemm_compute_slice(Alds[0], Blds[0], wm, wn, half, ln, acc);

        if (wv == 0) TDM_WAIT();            // buffer 1 slice landed
        __syncthreads();
        if (wv == 0 && kt + 64 < E_DIM) {   // prefetch into buffer 0
            tdm_load_2d((uint32_t)(uintptr_t)&Alds[0][0], Abase + kt + 64, E_DIM, MROWS, 32, 128, E_DIM);
            tdm_load_2d((uint32_t)(uintptr_t)&Blds[0][0], Bbase + kt + 64, E_DIM, E_DIM, 32, 128, E_DIM);
        }
        gemm_compute_slice(Alds[1], Blds[1], wm, wn, half, ln, acc);
    }
#else
    for (int kt = 0; kt < E_DIM; kt += 32) {
        __syncthreads();
        const int t = threadIdx.x, r = t >> 1, c = (t & 1) * 16;
        *(v16h*)&Alds[0][r * 32 + c] = *(const v16h*)(Abase + (size_t)r * E_DIM + kt + c);
        *(v16h*)&Blds[0][r * 32 + c] = *(const v16h*)(Bbase + (size_t)r * E_DIM + kt + c);
        __syncthreads();
        gemm_compute_slice(Alds[0], Blds[0], wm, wn, half, ln, acc);
    }
#endif

    // ---- epilogue ----
    #pragma unroll
    for (int mi = 0; mi < 2; ++mi) {
        #pragma unroll
        for (int nj = 0; nj < 4; ++nj) {
            const int   n  = n0 + wn * 64 + nj * 16 + ln;
            const float bn = bias[n];
            const int   h  = n >> 6;
            const int   d  = n & 63;
            v8f c = acc[mi][nj];
            if (mode == 3) {
                #pragma unroll
                for (int g = 0; g < 8; ++g) {
                    const int m = m0 + wm * 32 + mi * 16 + g + half * 8;
                    outf[(size_t)m * E_DIM + n] = c[g] + bn;
                }
            } else if (mode == 2) {
                #pragma unroll
                for (int g = 0; g < 8; ++g) {
                    const int m = m0 + wm * 32 + mi * 16 + g + half * 8;
                    const int s = m >> 1, b = m & 1;
                    outh[((size_t)(b * H_DIM + h) * HD + d) * S_DIM + s] =
                        (_Float16)(c[g] + bn);
                }
            } else {
                const float invf = __expf(-(float)(d & ~1) * (9.210340371976184f / 64.0f));
                const float qs   = (mode == 0) ? 0.125f : 1.0f;
                #pragma unroll
                for (int g = 0; g < 8; ++g) {
                    const int m = m0 + wm * 32 + mi * 16 + g + half * 8;
                    const int l = m >> 1, b = m & 1;
                    float x = c[g] + bn;
                    float p = __shfl_xor(x, 1);          // RoPE partner (d parity)
                    float sn, cs;
                    __sincosf((float)l * invf, &sn, &cs);
                    float r = (d & 1) ? (p * sn + x * cs) : (x * cs - p * sn);
                    outh[((size_t)(b * H_DIM + h) * L_DIM + l) * HD + d] =
                        (_Float16)(r * qs);
                }
            }
        }
    }
}

// ---------------------------------------------------------------------------
// Flash attention: 128-thread block = 64 query rows of one (b,h); the 4 waves
// share TDM-staged K/V S-chunks in LDS (double-buffered).
// ---------------------------------------------------------------------------
__device__ __forceinline__ void attn_stage(
    const _Float16* __restrict__ Kl,   // [32][HD] current S-chunk of K
    const _Float16* __restrict__ Vl,   // [HD][32] current S-chunk of V^T
    const uint8_t* __restrict__ mk, int s,
    v16h qa0, v16h qa1,
    _Float16 (* __restrict__ pl)[32],  // this wave's 16x32 P staging tile
    int half, int ln,
    v8f& o0, v8f& o1, v8f& o2, v8f& o3,
    float* __restrict__ mrow, float* __restrict__ lrow)
{
    v16h kb00 = *(const v16h*)&Kl[ln * HD + half * 16];
    v16h kb01 = *(const v16h*)&Kl[ln * HD + 32 + half * 16];
    v16h kb10 = *(const v16h*)&Kl[(16 + ln) * HD + half * 16];
    v16h kb11 = *(const v16h*)&Kl[(16 + ln) * HD + 32 + half * 16];
    v8f s0t = {}, s1t = {};
    s0t = __builtin_amdgcn_wmma_f32_16x16x32_f16(false, qa0, false, kb00, (short)0, s0t, false, false);
    s0t = __builtin_amdgcn_wmma_f32_16x16x32_f16(false, qa1, false, kb01, (short)0, s0t, false, false);
    s1t = __builtin_amdgcn_wmma_f32_16x16x32_f16(false, qa0, false, kb10, (short)0, s1t, false, false);
    s1t = __builtin_amdgcn_wmma_f32_16x16x32_f16(false, qa1, false, kb11, (short)0, s1t, false, false);

    const bool bad0 = mk[s + ln] != 0;
    const bool bad1 = mk[s + 16 + ln] != 0;

    #pragma unroll
    for (int g = 0; g < 8; ++g) {
        float e0 = bad0 ? -1e30f : s0t[g];
        float e1 = bad1 ? -1e30f : s1t[g];
        float nm = fmaxf(e0, e1);
        nm = fmaxf(nm, __shfl_xor(nm, 1));
        nm = fmaxf(nm, __shfl_xor(nm, 2));
        nm = fmaxf(nm, __shfl_xor(nm, 4));
        nm = fmaxf(nm, __shfl_xor(nm, 8));
        nm = fmaxf(nm, mrow[g]);
        const float corr = __expf(mrow[g] - nm);
        mrow[g] = nm;
        const float p0 = __expf(e0 - nm);
        const float p1 = __expf(e1 - nm);
        float rs = p0 + p1;
        rs += __shfl_xor(rs, 1);
        rs += __shfl_xor(rs, 2);
        rs += __shfl_xor(rs, 4);
        rs += __shfl_xor(rs, 8);
        lrow[g] = lrow[g] * corr + rs;
        o0[g] *= corr; o1[g] *= corr; o2[g] *= corr; o3[g] *= corr;
        const int r = g + half * 8;
        pl[r][ln]      = (_Float16)p0;
        pl[r][16 + ln] = (_Float16)p1;
    }

    const _Float16* pr = &pl[ln][half * 16];
    v16h pa = *(const v16h*)(pr);

    o0 = __builtin_amdgcn_wmma_f32_16x16x32_f16(false, pa, false,
             *(const v16h*)&Vl[(0  + ln) * 32 + half * 16], (short)0, o0, false, false);
    o1 = __builtin_amdgcn_wmma_f32_16x16x32_f16(false, pa, false,
             *(const v16h*)&Vl[(16 + ln) * 32 + half * 16], (short)0, o1, false, false);
    o2 = __builtin_amdgcn_wmma_f32_16x16x32_f16(false, pa, false,
             *(const v16h*)&Vl[(32 + ln) * 32 + half * 16], (short)0, o2, false, false);
    o3 = __builtin_amdgcn_wmma_f32_16x16x32_f16(false, pa, false,
             *(const v16h*)&Vl[(48 + ln) * 32 + half * 16], (short)0, o3, false, false);
}

__global__ __launch_bounds__(128) void attn_kernel(
    const _Float16* __restrict__ Qh,   // [BH, L, HD], pre-scaled by 1/8
    const _Float16* __restrict__ Kh,   // [BH, S, HD]
    const _Float16* __restrict__ Vt,   // [BH, HD, S]
    const uint8_t*  __restrict__ mask, // [B, S]
    _Float16* __restrict__ O)          // [MROWS, E] f16
{
    __shared__ __align__(64) _Float16 Klds[2][32 * HD];   // 2 x 4 KB
    __shared__ __align__(64) _Float16 Vlds[2][HD * 32];   // 2 x 4 KB
    __shared__ __align__(64) _Float16 plds[4][16][32];    // 4 KB

    const int lane = threadIdx.x & 31;
    const int wv   = threadIdx.x >> 5;
    const int bh   = blockIdx.x >> 5;             // 32 q-blocks of 64 rows per bh
    const int q0   = (blockIdx.x & 31) * 64 + wv * 16;
    const int b    = bh >> 4;
    const int h    = bh & 15;
    const int half = lane >> 4;
    const int ln   = lane & 15;

    const _Float16* Kbase = Kh + (size_t)bh * S_DIM * HD;
    const _Float16* Vbase = Vt + (size_t)bh * HD * S_DIM;
    const uint8_t*  mk    = mask + (size_t)b * S_DIM;

    const _Float16* qrow = Qh + ((size_t)bh * L_DIM + q0 + ln) * HD + half * 16;
    v16h qa0 = *(const v16h*)(qrow);
    v16h qa1 = *(const v16h*)(qrow + 32);

    v8f o0 = {}, o1 = {}, o2 = {}, o3 = {};
    float mrow[8], lrow[8];
    #pragma unroll
    for (int g = 0; g < 8; ++g) { mrow[g] = -1e30f; lrow[g] = 0.0f; }

#if USE_TDM
    if (wv == 0) {   // prologue: stage chunk s=0 into buffer 0
        tdm_load_2d((uint32_t)(uintptr_t)&Klds[0][0], Kbase, HD, S_DIM, HD, 32, HD);
        tdm_load_2d((uint32_t)(uintptr_t)&Vlds[0][0], Vbase, S_DIM, HD, 32, HD, S_DIM);
    }
    for (int s0 = 0; s0 < S_DIM; s0 += 64) {
        if (wv == 0) TDM_WAIT();
        __syncthreads();
        if (wv == 0) {   // prefetch chunk s0+32 into buffer 1
            tdm_load_2d((uint32_t)(uintptr_t)&Klds[1][0], Kbase + (size_t)(s0 + 32) * HD, HD, S_DIM, HD, 32, HD);
            tdm_load_2d((uint32_t)(uintptr_t)&Vlds[1][0], Vbase + s0 + 32, S_DIM, HD, 32, HD, S_DIM);
        }
        attn_stage(Klds[0], Vlds[0], mk, s0, qa0, qa1, plds[wv],
                   half, ln, o0, o1, o2, o3, mrow, lrow);

        if (wv == 0) TDM_WAIT();
        __syncthreads();
        if (wv == 0 && s0 + 64 < S_DIM) {   // prefetch chunk s0+64 into buffer 0
            tdm_load_2d((uint32_t)(uintptr_t)&Klds[0][0], Kbase + (size_t)(s0 + 64) * HD, HD, S_DIM, HD, 32, HD);
            tdm_load_2d((uint32_t)(uintptr_t)&Vlds[0][0], Vbase + s0 + 64, S_DIM, HD, 32, HD, S_DIM);
        }
        attn_stage(Klds[1], Vlds[1], mk, s0 + 32, qa0, qa1, plds[wv],
                   half, ln, o0, o1, o2, o3, mrow, lrow);
    }
#else
    for (int s0 = 0; s0 < S_DIM; s0 += 32) {
        __syncthreads();
        const int t = threadIdx.x;
        {   // cooperative copy: K chunk (32 x 64 halfs), V chunk (64 x 32 halfs)
            const int rk = t >> 2, ck = (t & 3) * 16;
            *(v16h*)&Klds[0][rk * HD + ck] = *(const v16h*)(Kbase + (size_t)(s0 + rk) * HD + ck);
            const int rv = t >> 1, cv = (t & 1) * 16;
            *(v16h*)&Vlds[0][rv * 32 + cv] = *(const v16h*)(Vbase + (size_t)rv * S_DIM + s0 + cv);
        }
        __syncthreads();
        attn_stage(Klds[0], Vlds[0], mk, s0, qa0, qa1, plds[wv],
                   half, ln, o0, o1, o2, o3, mrow, lrow);
    }
#endif

    #pragma unroll
    for (int g = 0; g < 8; ++g) {
        const float inv = 1.0f / lrow[g];
        const int l = q0 + g + half * 8;
        _Float16* orow = O + ((size_t)l * B_DIM + b) * E_DIM + h * 64;
        orow[ln]      = (_Float16)(o0[g] * inv);
        orow[16 + ln] = (_Float16)(o1[g] * inv);
        orow[32 + ln] = (_Float16)(o2[g] * inv);
        orow[48 + ln] = (_Float16)(o3[g] * inv);
    }
}

extern "C" void kernel_launch(void* const* d_in, const int* in_sizes, int n_in,
                              void* d_out, int out_size, void* d_ws, size_t ws_size,
                              hipStream_t stream) {
    const float*   query = (const float*)d_in[0];
    const float*   key   = (const float*)d_in[1];
    const float*   value = (const float*)d_in[2];
    const uint8_t* mask  = (const uint8_t*)d_in[3];
    const float*   Wq    = (const float*)d_in[4];
    const float*   bq    = (const float*)d_in[5];
    const float*   Wk    = (const float*)d_in[6];
    const float*   bk    = (const float*)d_in[7];
    const float*   Wv    = (const float*)d_in[8];
    const float*   bv    = (const float*)d_in[9];
    const float*   Wo    = (const float*)d_in[10];
    const float*   bo    = (const float*)d_in[11];

    char* ws = (char*)d_ws;
    const size_t MB = 1024 * 1024;
    _Float16* Xq = (_Float16*)(ws);              // 8 MB  f16 query
    _Float16* Xk = (_Float16*)(ws + 8  * MB);    // 8 MB  f16 key
    _Float16* Xv = (_Float16*)(ws + 16 * MB);    // 8 MB  f16 value
    _Float16* Hq = (_Float16*)(ws + 24 * MB);    // 2 MB  f16 Wq
    _Float16* Hk = (_Float16*)(ws + 26 * MB);    // 2 MB  f16 Wk
    _Float16* Hv = (_Float16*)(ws + 28 * MB);    // 2 MB  f16 Wv
    _Float16* Ho = (_Float16*)(ws + 30 * MB);    // 2 MB  f16 Wo
    _Float16* Qh = (_Float16*)(ws + 32 * MB);    // 8 MB  [bh][l][d]
    _Float16* Kh = (_Float16*)(ws + 40 * MB);    // 8 MB  [bh][s][d]
    _Float16* Vt = (_Float16*)(ws + 48 * MB);    // 8 MB  [bh][d][s]
    _Float16* Oh = (_Float16*)(ws + 56 * MB);    // 8 MB  attention out f16
    float*    out = (float*)d_out;

    dim3 b256(256);
    cvt_f32_f16_kernel<<<dim3(2048), b256, 0, stream>>>(query, Xq);
    cvt_f32_f16_kernel<<<dim3(2048), b256, 0, stream>>>(key,   Xk);
    cvt_f32_f16_kernel<<<dim3(2048), b256, 0, stream>>>(value, Xv);
    cvt_f32_f16_kernel<<<dim3(512),  b256, 0, stream>>>(Wq, Hq);
    cvt_f32_f16_kernel<<<dim3(512),  b256, 0, stream>>>(Wk, Hk);
    cvt_f32_f16_kernel<<<dim3(512),  b256, 0, stream>>>(Wv, Hv);
    cvt_f32_f16_kernel<<<dim3(512),  b256, 0, stream>>>(Wo, Ho);

    dim3 ggemm(MROWS / 128, E_DIM / 128);        // 32 x 8 blocks
    gemm_kernel<<<ggemm, b256, 0, stream>>>(Xq, Hq, bq, Qh, nullptr, 0);
    gemm_kernel<<<ggemm, b256, 0, stream>>>(Xk, Hk, bk, Kh, nullptr, 1);
    gemm_kernel<<<ggemm, b256, 0, stream>>>(Xv, Hv, bv, Vt, nullptr, 2);

    attn_kernel<<<dim3(1024), dim3(128), 0, stream>>>(Qh, Kh, Vt, mask, Oh);

    gemm_kernel<<<ggemm, b256, 0, stream>>>(Oh, Ho, bo, nullptr, out, 3);
}
